// S4D_51951924412417
// MI455X (gfx1250) — compile-verified, matrix operand order checked by reading, not run on hardware
//
#include <hip/hip_runtime.h>
#include <hip/hip_bf16.h>
#include <math.h>

// Problem constants (B, H, N, L) = (8, 1024, 64, 2048)
constexpr int cB  = 8;
constexpr int cH  = 1024;
constexpr int cNH = 32;          // N/2 complex modes
constexpr int cL  = 2048;
constexpr int cNBL = cB * cL;    // 16384 columns of Y

typedef __attribute__((ext_vector_type(16))) __bf16 v16bf;
typedef __attribute__((ext_vector_type(8)))  float  v8f;
typedef __attribute__((ext_vector_type(4)))  int    v4i;

#define AS1 __attribute__((address_space(1)))
#define AS3 __attribute__((address_space(3)))

#if __has_builtin(__builtin_amdgcn_global_load_async_to_lds_b128) && \
    __has_builtin(__builtin_amdgcn_s_wait_asynccnt)
#define USE_ASYNC_LDS 1
#else
#define USE_ASYNC_LDS 0
#endif

// ---------------------------------------------------------------------------
// Kernel 1: synthesize S4D convolution kernel K[h,l] in bf16.
// K[h,l] = 2 * sum_n Re( Cc[h,n] * exp(dtA[h,n] * l) )
// ---------------------------------------------------------------------------
__global__ void s4d_kernel_k(const float* __restrict__ C,
                             const float* __restrict__ log_dt,
                             const float* __restrict__ log_A_real,
                             const float* __restrict__ A_imag,
                             __bf16* __restrict__ Kbf) {
  __shared__ float ar[cNH], ai[cNH], cr[cNH], ci[cNH];
  const int h = blockIdx.x;
  const int tid = threadIdx.x;
  if (tid < cNH) {
    const int n = tid;
    const float dt  = __expf(log_dt[h]);
    const float Are = -__expf(log_A_real[h * cNH + n]);
    const float Aim = A_imag[h * cNH + n];
    const float dtr = Are * dt, dti = Aim * dt;
    const float e = __expf(dtr);
    float sn, cs;
    __sincosf(dti, &sn, &cs);
    const float wr = e * cs - 1.0f, wi = e * sn;        // exp(dtA) - 1
    const float c0 = C[(h * cNH + n) * 2 + 0];
    const float c1 = C[(h * cNH + n) * 2 + 1];
    const float tr = c0 * wr - c1 * wi;
    const float ti = c0 * wi + c1 * wr;
    const float inv = 1.0f / (Are * Are + Aim * Aim);
    cr[n] = (tr * Are + ti * Aim) * inv;                 // * conj(A)/|A|^2
    ci[n] = (ti * Are - tr * Aim) * inv;
    ar[n] = dtr;
    ai[n] = dti;
  }
  __syncthreads();
  for (int l = tid; l < cL; l += blockDim.x) {
    const float lf = (float)l;
    float s = 0.0f;
#pragma unroll 4
    for (int n = 0; n < cNH; ++n) {
      const float e = __expf(ar[n] * lf);
      float sn, cs;
      sincosf(ai[n] * lf, &sn, &cs);                     // large args: precise
      s += e * (cr[n] * cs - ci[n] * sn);
    }
    Kbf[h * cL + l] = (__bf16)(2.0f * s);
  }
}

// ---------------------------------------------------------------------------
// Kernel 2: W (2H x H fp32) -> bf16
// ---------------------------------------------------------------------------
__global__ void w_to_bf16(const float* __restrict__ W,
                          __bf16* __restrict__ Wbf, int n) {
  const int i = blockIdx.x * blockDim.x + threadIdx.x;
  if (i < n) Wbf[i] = (__bf16)W[i];
}

// ---------------------------------------------------------------------------
// Kernel 3: causal convolution as block-Toeplitz WMMA, fused +u*D and GELU.
// Block = (h, 256-l segment); 8 waves, each owning TWO 16(l) x 16(b) fp32
// tiles so one staged u-tile feeds 2 WMMAs. Cols 0..7 = batches, 8..15 pad.
// A fragment (16x32 bf16, ISA 7.12.2): lane -> M=lane%16, half=lane/16,
//   element j -> K = (j/8)*16 + half*8 + (j%8).  B mirrored with N=lane%16.
// ---------------------------------------------------------------------------
__global__ void conv_wmma(const float* __restrict__ u,
                          const __bf16* __restrict__ Kbf,
                          const float* __restrict__ D,
                          __bf16* __restrict__ Ybf) {
  __shared__ __bf16 kl[288];        // sliding Toeplitz window for 256 l's
  __shared__ __bf16 ub[16 * 32];    // u tile, [n][k] with n = batch (pad to 16)
  const int h      = blockIdx.y;
  const int l_base = blockIdx.x * 256;
  const int tid    = threadIdx.x;
  const int wv     = tid >> 5;
  const int lane   = tid & 31;
  const int half   = lane >> 4;
  const int row    = lane & 15;
  const int l0a    = l_base + wv * 16;        // first tile
  const int l0b    = l0a + 128;               // second tile

  v8f acca = {}, accb = {};
  const int nchunk = (l_base >> 5) + 8;       // causality: m <= l_base + 255
  for (int c = 0; c < nchunk; ++c) {
    const int m0   = c << 5;
    const int dmin = l_base - m0 - 31;
    for (int i = tid; i < 288; i += 256) {
      const int d = dmin + i;                 // d = l - m
      kl[i] = (d >= 0 && d < cL) ? Kbf[h * cL + d] : (__bf16)0.0f;
    }
    for (int i = tid; i < 512; i += 256) {
      const int n = i >> 5, kk = i & 31;
      const float v = (n < cB) ? u[((n * cH) + h) * cL + (m0 + kk)] : 0.0f;
      ub[i] = (__bf16)v;
    }
    __syncthreads();
    v16bf aa, ab, bb;
#pragma unroll
    for (int j = 0; j < 16; ++j) {
      const int kk = ((j >> 3) << 4) + half * 8 + (j & 7);
      aa[j] = kl[wv * 16 + row - kk + 31];         // K[h, l0a+row-(m0+kk)]
      ab[j] = kl[128 + wv * 16 + row - kk + 31];   // K[h, l0b+row-(m0+kk)]
      bb[j] = ub[row * 32 + kk];                   // u[b=row, m0+kk]
    }
    // Wave-uniform causality guards (readfirstlane keeps EXEC all-1s for WMMA)
    if (__builtin_amdgcn_readfirstlane((int)(m0 <= l0a + 15)))
      acca = __builtin_amdgcn_wmma_f32_16x16x32_bf16(
          false, aa, false, bb, (short)0, acca, false, false);
    if (__builtin_amdgcn_readfirstlane((int)(m0 <= l0b + 15)))
      accb = __builtin_amdgcn_wmma_f32_16x16x32_bf16(
          false, ab, false, bb, (short)0, accb, false, false);
    __syncthreads();
  }

  // Epilogue: C layout -> lane N=lane%16 (batch), element r -> M = half*8 + r.
  const int bcol = row;
  if (bcol < cB) {
    const float Dh = D[h];
#pragma unroll
    for (int r = 0; r < 8; ++r) {
      const int la = l0a + half * 8 + r;
      float ya = acca[r] + u[((bcol * cH) + h) * cL + la] * Dh;
      ya = 0.5f * ya * (1.0f + erff(ya * 0.70710678f));
      Ybf[h * cNBL + bcol * cL + la] = (__bf16)ya;

      const int lb = l0b + half * 8 + r;
      float yb = accb[r] + u[((bcol * cH) + h) * cL + lb] * Dh;
      yb = 0.5f * yb * (1.0f + erff(yb * 0.70710678f));
      Ybf[h * cNBL + bcol * cL + lb] = (__bf16)yb;
    }
  }
}

// ---------------------------------------------------------------------------
// Kernel 4: z = W_bf16 (2048x1024) @ Y_bf16 (1024x16384), fused bias + gate.
// Block = 32 o'-rows x 128 n-cols. Each wave: one n-tile, FOUR accumulators
// (a_lo/a_hi paired with g_lo/g_hi rows +1024) -> 4 WMMAs share one B
// fragment per K-chunk. B tile staged [k][n] via async global->LDS b128
// copies (ASYNCcnt) when the toolchain exposes the builtin.
// ---------------------------------------------------------------------------
__global__ void gemm_gate(const __bf16* __restrict__ Wbf,
                          const __bf16* __restrict__ Ybf,
                          const float* __restrict__ bias,
                          float* __restrict__ out) {
  __shared__ __bf16 As[4][16 * 32];   // 0:a_lo 1:a_hi 2:g_lo 3:g_hi
  __shared__ __bf16 Bs[32 * 136];     // [k][n], rows padded 128->136 (16B align)
  const int nblock = blockIdx.x * 128;
  const int o_base = blockIdx.y * 32;
  const int tid  = threadIdx.x;
  const int wv   = tid >> 5;
  const int lane = tid & 31;
  const int half = lane >> 4;
  const int row  = lane & 15;

  v8f accAlo = {}, accAhi = {}, accGlo = {}, accGhi = {};
  for (int k0 = 0; k0 < cH; k0 += 32) {
    // ---- stage W tiles (4 x 16x32) ----
    for (int i = tid; i < 2048; i += 256) {
      const int t = i >> 9;                       // which of the 4 tiles
      const int r = (i >> 5) & 15, kk = i & 31;
      const int orow = o_base + (t & 1) * 16 + (t >> 1) * cH + r;
      As[t][r * 32 + kk] = Wbf[orow * cH + k0 + kk];
    }
    // ---- stage Y tile (32 k x 128 n) ----
#if USE_ASYNC_LDS
#pragma unroll
    for (int it = 0; it < 2; ++it) {
      const int i  = tid + it * 256;              // 0..511: one b128 each
      const int kk = i >> 4;
      const int n8 = (i & 15) * 8;                // 8 bf16 = 16 bytes
      const __bf16* g = &Ybf[(size_t)(k0 + kk) * cNBL + nblock + n8];
      AS1 v4i* gp = (AS1 v4i*)(uintptr_t)g;
      AS3 v4i* lp = (AS3 v4i*)(unsigned)(uintptr_t)(&Bs[kk * 136 + n8]);
      __builtin_amdgcn_global_load_async_to_lds_b128(gp, lp, 0, 0);
    }
    __builtin_amdgcn_s_wait_asynccnt(0);
#else
    for (int i = tid; i < 4096; i += 256) {       // coalesced along n
      const int kk = i >> 7, nl = i & 127;
      Bs[kk * 136 + nl] = Ybf[(k0 + kk) * cNBL + nblock + nl];
    }
#endif
    if (k0 + 32 < cH)                             // prefetch next Y chunk
      __builtin_prefetch(&Ybf[(k0 + 32) * cNBL + nblock + (tid & 127)], 0, 0);
    __syncthreads();

    v16bf a0, a1, a2, a3, bb;
#pragma unroll
    for (int j = 0; j < 16; ++j) {
      const int kk = ((j >> 3) << 4) + half * 8 + (j & 7);
      a0[j] = As[0][row * 32 + kk];
      a1[j] = As[1][row * 32 + kk];
      a2[j] = As[2][row * 32 + kk];
      a3[j] = As[3][row * 32 + kk];
      bb[j] = Bs[kk * 136 + wv * 16 + row];       // N = wave n-tile + lane
    }
    accAlo = __builtin_amdgcn_wmma_f32_16x16x32_bf16(
        false, a0, false, bb, (short)0, accAlo, false, false);
    accAhi = __builtin_amdgcn_wmma_f32_16x16x32_bf16(
        false, a1, false, bb, (short)0, accAhi, false, false);
    accGlo = __builtin_amdgcn_wmma_f32_16x16x32_bf16(
        false, a2, false, bb, (short)0, accGlo, false, false);
    accGhi = __builtin_amdgcn_wmma_f32_16x16x32_bf16(
        false, a3, false, bb, (short)0, accGhi, false, false);
    __syncthreads();
  }

  const int nl = wv * 16 + row;
  const int n  = nblock + nl;
  const int b  = n >> 11;          // n / L
  const int l  = n & (cL - 1);
#pragma unroll
  for (int r = 0; r < 8; ++r) {
    const int mrow = half * 8 + r;
    {
      const int oo = o_base + mrow;
      const float aa = accAlo[r] + bias[oo];
      const float gg = accGlo[r] + bias[oo + cH];
      out[((b * cH) + oo) * cL + l] = aa * (1.0f / (1.0f + __expf(-gg)));
    }
    {
      const int oo = o_base + 16 + mrow;
      const float aa = accAhi[r] + bias[oo];
      const float gg = accGhi[r] + bias[oo + cH];
      out[((b * cH) + oo) * cL + l] = aa * (1.0f / (1.0f + __expf(-gg)));
    }
  }
}

// ---------------------------------------------------------------------------
extern "C" void kernel_launch(void* const* d_in, const int* in_sizes, int n_in,
                              void* d_out, int out_size, void* d_ws, size_t ws_size,
                              hipStream_t stream) {
  const float* u          = (const float*)d_in[0];
  const float* C          = (const float*)d_in[1];
  const float* log_dt     = (const float*)d_in[2];
  const float* log_A_real = (const float*)d_in[3];
  const float* A_imag     = (const float*)d_in[4];
  const float* D          = (const float*)d_in[5];
  const float* W          = (const float*)d_in[6];
  const float* bias       = (const float*)d_in[7];
  float* out = (float*)d_out;

  char* ws = (char*)d_ws;
  __bf16* Kbf = (__bf16*)(ws);                            //  4 MB: H*L
  __bf16* Wbf = (__bf16*)(ws + ((size_t)4 << 20));        //  4 MB: 2H*H
  __bf16* Ybf = (__bf16*)(ws + ((size_t)8 << 20));        // 32 MB: H*(B*L)

  s4d_kernel_k<<<cH, 256, 0, stream>>>(C, log_dt, log_A_real, A_imag, Kbf);
  w_to_bf16<<<(2 * cH * cH + 255) / 256, 256, 0, stream>>>(W, Wbf, 2 * cH * cH);
  conv_wmma<<<dim3(8, cH), 256, 0, stream>>>(u, Kbf, D, Ybf);
  gemm_gate<<<dim3(128, 32), 256, 0, stream>>>(Wbf, Ybf, bias, out);
}
